// SelfAttention_88862873354940
// MI455X (gfx1250) — matrix operand order repaired and checked by reading, not compiled, not run on hardware
//
#include <hip/hip_runtime.h>

// Fused flash-attention: s = g@f^T ; beta = softmax(s) ; out = gamma*(beta@h) + img
// B=8, N=4096, C=128, Cr=16. fp32 in/out.
// Pass 1 (prep): convert g (x log2e), f (transposed), h to bf16 in d_ws.
// Pass 2 (fa):   flash attention with v_wmma_f32_16x16x32_bf16, exp2-domain softmax,
//                double-buffered async global->LDS staging (ASYNCcnt), permlane16
//                butterfly reductions.

#define NTOK  4096
#define CCH   128
#define CRD   16
#define KT    64      // keys per LDS tile
#define ROWF  72      // padded row (shorts) of F^T tile  [32 c][64 keys]
#define ROWV  136     // padded row (shorts) of V tile    [64 keys][128 c]
#define ROWP  72      // padded row (shorts) of per-wave P scratch [16 q][64 keys]

typedef __attribute__((ext_vector_type(16))) __bf16       v16bf;
typedef __attribute__((ext_vector_type(8)))  float        v8f;
typedef __attribute__((ext_vector_type(8)))  unsigned int v8u;
typedef __attribute__((ext_vector_type(4)))  int          v4i;

#if defined(__has_builtin)
#if __has_builtin(__builtin_amdgcn_global_load_async_to_lds_b128)
#define HAS_ASYNC 1
#endif
#if __has_builtin(__builtin_amdgcn_permlane16)
#define PERMLANE16(s, a, b) __builtin_amdgcn_permlane16((s), (s), (a), (b), false, false)
#elif __has_builtin(__builtin_amdgcn_permlane16_b32)
#define PERMLANE16(s, a, b) __builtin_amdgcn_permlane16_b32((s), (s), (a), (b), false, false)
#endif
#endif
#ifndef HAS_ASYNC
#define HAS_ASYNC 0
#endif

static __device__ __forceinline__ void cp_async16(const void* g, void* l) {
#if HAS_ASYNC
  __builtin_amdgcn_global_load_async_to_lds_b128((v4i*)g, (v4i*)l, 0, 0);
#else
  *(uint4*)l = *(const uint4*)g;
#endif
}
static __device__ __forceinline__ void async_wait() {
#if HAS_ASYNC
#if defined(__has_builtin) && __has_builtin(__builtin_amdgcn_s_wait_asynccnt)
  __builtin_amdgcn_s_wait_asynccnt(0);
#else
  __asm__ __volatile__("s_wait_asynccnt 0" ::: "memory");
#endif
#endif
}

// butterfly reductions over each 16-lane group (broadcast result to all 16 lanes)
#ifdef PERMLANE16
static __device__ __forceinline__ float pl16f(float v, unsigned s1, unsigned s2) {
  return __uint_as_float(PERMLANE16(__float_as_uint(v), s1, s2));
}
static __device__ __forceinline__ float redmax16(float v) {
  v = fmaxf(v, pl16f(v, 0x67452301u, 0xEFCDAB89u));   // xor 1
  v = fmaxf(v, pl16f(v, 0x54761032u, 0xDCFE98BAu));   // xor 2
  v = fmaxf(v, pl16f(v, 0x32107654u, 0xBA98FEDCu));   // xor 4
  v = fmaxf(v, pl16f(v, 0xFEDCBA98u, 0x76543210u));   // xor 8
  return v;
}
static __device__ __forceinline__ float redsum16(float v) {
  v += pl16f(v, 0x67452301u, 0xEFCDAB89u);
  v += pl16f(v, 0x54761032u, 0xDCFE98BAu);
  v += pl16f(v, 0x32107654u, 0xBA98FEDCu);
  v += pl16f(v, 0xFEDCBA98u, 0x76543210u);
  return v;
}
#else
static __device__ __forceinline__ float redmax16(float v) {
  v = fmaxf(v, __shfl_xor(v, 1)); v = fmaxf(v, __shfl_xor(v, 2));
  v = fmaxf(v, __shfl_xor(v, 4)); v = fmaxf(v, __shfl_xor(v, 8));
  return v;
}
static __device__ __forceinline__ float redsum16(float v) {
  v += __shfl_xor(v, 1); v += __shfl_xor(v, 2);
  v += __shfl_xor(v, 4); v += __shfl_xor(v, 8);
  return v;
}
#endif

static __device__ __forceinline__ unsigned short f2bf(float x) {   // RNE (cold path)
  unsigned u = __float_as_uint(x);
  u += 0x7FFFu + ((u >> 16) & 1u);
  return (unsigned short)(u >> 16);
}
static __device__ __forceinline__ unsigned pk2(float lo, float hi) {
  return (unsigned)f2bf(lo) | ((unsigned)f2bf(hi) << 16);
}
static __device__ __forceinline__ unsigned short f2bf_fast(float x) { // probs in [0,1]
  return (unsigned short)((__float_as_uint(x) + 0x8000u) >> 16);
}
static __device__ __forceinline__ v16bf mk16(uint4 a, uint4 b) {
  v8u u;
  u[0]=a.x; u[1]=a.y; u[2]=a.z; u[3]=a.w;
  u[4]=b.x; u[5]=b.y; u[6]=b.z; u[7]=b.w;
  return __builtin_bit_cast(v16bf, u);
}

// ---------------- pass 1: bf16 pre-conversion ----------------
#define NH4 1048576   // 8*4096*128/4  float4 units of h
#define NG4  131072   // 8*4096*16/4   float4 units of g (and of f)

__global__ __launch_bounds__(256) void prep_kernel(
    const float* __restrict__ f, const float* __restrict__ g,
    const float* __restrict__ h, unsigned short* __restrict__ Hbf,
    unsigned short* __restrict__ FT, unsigned short* __restrict__ Qbf)
{
  const int u = blockIdx.x * 256 + threadIdx.x;
  if (u < NH4) {                                   // h -> bf16, same layout
    float4 v = ((const float4*)h)[u];
    ((uint2*)Hbf)[u] = make_uint2(pk2(v.x, v.y), pk2(v.z, v.w));
  } else if (u < NH4 + NG4) {                      // g * log2(e) -> bf16
    const int i = u - NH4;
    const float s = 1.4426950408889634f;
    float4 v = ((const float4*)g)[i];
    ((uint2*)Qbf)[i] = make_uint2(pk2(v.x * s, v.y * s), pk2(v.z * s, v.w * s));
  } else {                                         // f -> bf16 transposed [b][c][n]
    const int i  = u - NH4 - NG4;
    float4 v = ((const float4*)f)[i];
    const int cg = (i & 3) << 2;
    const int bn = i >> 2;                         // b*4096 + n
    const int b  = bn >> 12;
    const int n  = bn & 4095;
    unsigned short* p = FT + ((size_t)(b * 16 + cg)) * NTOK + n;
    p[0 * NTOK] = f2bf(v.x);
    p[1 * NTOK] = f2bf(v.y);
    p[2 * NTOK] = f2bf(v.z);
    p[3 * NTOK] = f2bf(v.w);
  }
}

// ---------------- pass 2: flash attention ----------------
__global__ __launch_bounds__(256) void fa_kernel(
    const float* __restrict__ img,
    const unsigned short* __restrict__ Qbf,   // [8][4096][16]  bf16, pre-scaled log2e
    const unsigned short* __restrict__ FT,    // [8][16][4096]  bf16
    const unsigned short* __restrict__ Hbf,   // [8][4096][128] bf16
    const float* __restrict__ gamma, float* __restrict__ out)
{
  __shared__ __align__(16) unsigned short sF[2][32][ROWF];    // F^T tiles (dbl buf)
  __shared__ __align__(16) unsigned short sV[2][KT][ROWV];    // V tiles  (dbl buf)
  __shared__ __align__(16) unsigned short sP[8][16][ROWP];    // per-wave P scratch

  const int tid  = threadIdx.x;
  const int lane = tid & 31;
  const int wave = tid >> 5;
  const int half = lane >> 4;
  const int l16  = lane & 15;

  const int batch = blockIdx.x >> 5;
  const int q0    = ((blockIdx.x & 31) << 7) + (wave << 4);

  // zero only the K-padding rows (c=16..31) of both F buffers; never overwritten
  for (int i = tid; i < (16 * ROWF) / 2; i += 256) {
    ((unsigned*)&sF[0][16][0])[i] = 0u;
    ((unsigned*)&sF[1][16][0])[i] = 0u;
  }

  const int key = tid >> 2;     // 4 threads per key for V staging
  const int seg = tid & 3;

  auto stage = [&](int buf, int kt) {
    if (tid < 128) {            // F^T tile: 16 rows x 128B, one b128 per thread
      const int c = tid >> 3, s8 = tid & 7;
      cp_async16(FT + ((size_t)(batch * 16 + c)) * NTOK + kt + s8 * 8,
                 &sF[buf][c][s8 * 8]);
    }
    const unsigned short* src =   // V tile: 64 rows x 256B, 4 b128 per thread
        Hbf + ((size_t)(batch * NTOK + kt + key)) * CCH + seg * 32;
    #pragma unroll
    for (int i = 0; i < 4; ++i)
      cp_async16(src + i * 8, &sV[buf][key][seg * 32 + i * 8]);
  };

  // Q as bf16 A-operand (16x32, K=16..31 zero)
  v16bf qA;
  {
    const uint4 qv = *(const uint4*)(Qbf + ((size_t)(batch * NTOK + q0 + l16)) * CRD + half * 8);
    v8u u;
    u[0] = qv.x; u[1] = qv.y; u[2] = qv.z; u[3] = qv.w;
    u[4] = 0u;   u[5] = 0u;   u[6] = 0u;   u[7] = 0u;
    qA = __builtin_bit_cast(v16bf, u);
  }

  v8f acc[8];
  float mrow[8], lrow[8];
  #pragma unroll
  for (int j = 0; j < 8; ++j)
    #pragma unroll
    for (int r = 0; r < 8; ++r) acc[j][r] = 0.f;
  #pragma unroll
  for (int r = 0; r < 8; ++r) { mrow[r] = -3.0e38f; lrow[r] = 0.f; }

  stage(0, 0);                  // prologue: prefetch tile 0

  int it = 0;
  for (int kt = 0; kt < NTOK; kt += KT, ++it) {
    const int buf = it & 1;
    async_wait();               // tile `it` (issued one iteration ago) landed
    __syncthreads();            // all waves done staging `it` + done reading buf^1
    if (kt + KT < NTOK) stage(buf ^ 1, kt + KT);   // prefetch next tile

    // ---- scores (log2 domain): 4 WMMA C-tiles of 16q x 16k
    v8f s[4];
    #pragma unroll
    for (int ks = 0; ks < 4; ++ks) {
      v16bf Bm;
      #pragma unroll
      for (int e = 0; e < 16; ++e)
        Bm[e] = __builtin_bit_cast(__bf16, sF[buf][(e < 8 ? (half << 3) + e : 16 + (half << 3) + e - 8)][ks * 16 + l16]);
      v8f z = {0.f, 0.f, 0.f, 0.f, 0.f, 0.f, 0.f, 0.f};
      s[ks] = __builtin_amdgcn_wmma_f32_16x16x32_bf16(false, qA, false, Bm,
                                                      (short)0, z, false, false);
    }

    // ---- online softmax in exp2 domain (permlane16 butterflies)
    #pragma unroll
    for (int r = 0; r < 8; ++r) {
      float mx = fmaxf(fmaxf(s[0][r], s[1][r]), fmaxf(s[2][r], s[3][r]));
      mx = redmax16(mx);
      const float mnew = fmaxf(mrow[r], mx);
      const float sc   = __builtin_amdgcn_exp2f(mrow[r] - mnew);
      mrow[r] = mnew;
      float rs = 0.f;
      #pragma unroll
      for (int ks = 0; ks < 4; ++ks) {
        float p = __builtin_amdgcn_exp2f(s[ks][r] - mnew);
        s[ks][r] = p;
        rs += p;
      }
      rs = redsum16(rs);
      lrow[r] = lrow[r] * sc + rs;
      #pragma unroll
      for (int j = 0; j < 8; ++j) acc[j][r] *= sc;
    }

    // ---- P: C-layout -> A-layout via wave-private LDS round trip
    #pragma unroll
    for (int ks = 0; ks < 4; ++ks)
      #pragma unroll
      for (int r = 0; r < 8; ++r)
        sP[wave][(half << 3) + r][ks * 16 + l16] = f2bf_fast(s[ks][r]);

    __asm__ __volatile__("s_wait_dscnt 0" ::: "memory");  // LDS is in-order per wave

    v16bf pA0, pA1;
    {
      const char* base = (const char*)&sP[wave][l16][0] + (half << 4);
      uint4 a0 = *(const uint4*)(base +  0);
      uint4 a1 = *(const uint4*)(base + 32);
      uint4 a2 = *(const uint4*)(base + 64);
      uint4 a3 = *(const uint4*)(base + 96);
      pA0 = mk16(a0, a1);
      pA1 = mk16(a2, a3);
    }

    // ---- O += P * V : 8 column tiles x 2 K-subtiles
    #pragma unroll
    for (int j = 0; j < 8; ++j) {
      {
        v16bf Vb;
        #pragma unroll
        for (int e = 0; e < 16; ++e)
          Vb[e] = __builtin_bit_cast(__bf16, sV[buf][(e < 8 ? (half << 3) + e : 16 + (half << 3) + e - 8)][j * 16 + l16]);
        acc[j] = __builtin_amdgcn_wmma_f32_16x16x32_bf16(false, pA0, false,
                     Vb, (short)0, acc[j], false, false);
      }
      {
        v16bf Vb;
        #pragma unroll
        for (int e = 0; e < 16; ++e)
          Vb[e] = __builtin_bit_cast(__bf16, sV[buf][32 + (e < 8 ? (half << 3) + e : 16 + (half << 3) + e - 8)][j * 16 + l16]);
        acc[j] = __builtin_amdgcn_wmma_f32_16x16x32_bf16(false, pA1, false,
                     Vb, (short)0, acc[j], false, false);
      }
    }
  }

  // ---- epilogue: out = gamma * (O / l) + img
  const float gam = gamma[0];
  #pragma unroll
  for (int j = 0; j < 8; ++j) {
    #pragma unroll
    for (int r = 0; r < 8; ++r) {
      const int row = q0 + (half << 3) + r;
      const size_t idx = ((size_t)(batch * NTOK + row)) * CCH + j * 16 + l16;
      out[idx] = gam * (acc[j][r] / lrow[r]) + img[idx];
    }
  }
}

extern "C" void kernel_launch(void* const* d_in, const int* in_sizes, int n_in,
                              void* d_out, int out_size, void* d_ws, size_t ws_size,
                              hipStream_t stream) {
  const float* img   = (const float*)d_in[0];
  const float* f     = (const float*)d_in[1];
  const float* g     = (const float*)d_in[2];
  const float* h     = (const float*)d_in[3];
  const float* gamma = (const float*)d_in[4];
  float* out = (float*)d_out;
  (void)in_sizes; (void)n_in; (void)out_size; (void)ws_size;

  // workspace layout (bf16): Hbf 8MB | FT 1MB | Qbf 1MB  (10 MB total)
  unsigned short* Hbf = (unsigned short*)d_ws;
  unsigned short* FT  = Hbf + (size_t)8 * 4096 * 128;
  unsigned short* Qbf = FT  + (size_t)8 * 16 * 4096;

  prep_kernel<<<dim3((NH4 + 2 * NG4) / 256), dim3(256), 0, stream>>>(f, g, h, Hbf, FT, Qbf);
  fa_kernel<<<dim3(256), dim3(256), 0, stream>>>(img, Qbf, FT, Hbf, gamma, out);
}
